// GCN_85134841741499
// MI455X (gfx1250) — compile-verified, matrix operand order checked by reading, not run on hardware
//
#include <hip/hip_runtime.h>
#include <cstddef>

#define DF 128

typedef __attribute__((ext_vector_type(16))) __bf16 v16bf;
typedef __attribute__((ext_vector_type(8)))  __bf16 v8bf;
typedef __attribute__((ext_vector_type(4)))  __bf16 v4bf;
typedef __attribute__((ext_vector_type(8)))  float  v8f;

// ---------------------------------------------------------------------------
// Degree / normalization precompute (done once; shared by all 3 GCN layers)
// ---------------------------------------------------------------------------
__global__ void gcn_deg_init(float* __restrict__ deg, int n) {
    int i = blockIdx.x * blockDim.x + threadIdx.x;
    if (i < n) deg[i] = 1.0f;  // self-loop weight
}

__global__ void gcn_deg_accum(float* __restrict__ deg, const int* __restrict__ dst,
                              const float* __restrict__ ew, int ne) {
    int e = blockIdx.x * blockDim.x + threadIdx.x;
    if (e < ne) unsafeAtomicAdd(&deg[dst[e]], ew[e]);
}

__global__ void gcn_deg_finalize(float* __restrict__ deg, int n) {
    int i = blockIdx.x * blockDim.x + threadIdx.x;
    if (i < n) {
        float d = deg[i];
        deg[i] = (d > 0.0f) ? rsqrtf(d) : 0.0f;  // in-place -> dinv
    }
}

__global__ void gcn_edge_norm(float* __restrict__ norm, const float* __restrict__ dinv,
                              const int* __restrict__ src, const int* __restrict__ dst,
                              const float* __restrict__ ew, int ne) {
    int e = blockIdx.x * blockDim.x + threadIdx.x;
    if (e < ne) norm[e] = dinv[src[e]] * ew[e] * dinv[dst[e]];
}

// ---------------------------------------------------------------------------
// Embedding gather -> bf16 activations
// ---------------------------------------------------------------------------
__global__ void gcn_embed(__bf16* __restrict__ h0, const float* __restrict__ emb,
                          const int* __restrict__ x, long long n) {
    long long i = (long long)blockIdx.x * blockDim.x + threadIdx.x;
    if (i >= n) return;
    long long node = i >> 7;
    int f = (int)(i & (DF - 1));
    h0[i] = (__bf16)emb[(size_t)x[node] * DF + f];
}

// ---------------------------------------------------------------------------
// WMMA GEMM: out = h_in(bf16) @ W(fp32->bf16 in LDS).T
//  mode 0: write hw (bf16 messages) and seed agg = acc * dinv^2 (self loop)
//  mode 1: write fout(fp32) = acc + bias   (final output layer)
// Block = 256 threads = 8 waves; block covers 128 rows x 128 cols.
// Each wave: 16 rows, 8 column tiles, 4 K-steps of 32 -> 32 v_wmma per wave.
// ---------------------------------------------------------------------------
__global__ __launch_bounds__(256)
void gcn_gemm(const __bf16* __restrict__ hin, const float* __restrict__ Wf,
              const float* __restrict__ dinv, const float* __restrict__ bias,
              __bf16* __restrict__ hw, float* __restrict__ fout,
              int nrows, int mode) {
    __shared__ alignas(16) __bf16 Ws[DF * DF];  // 32 KB: W row-major as bf16
    const int tid = threadIdx.x;
    #pragma unroll 4
    for (int i = tid; i < DF * DF; i += 256) Ws[i] = (__bf16)Wf[i];
    __syncthreads();

    const int wave = tid >> 5;
    const int lane = tid & 31;
    const int hl   = lane >> 4;   // which 16-lane half
    const int r    = lane & 15;
    const size_t m0 = (size_t)blockIdx.x * 128 + (size_t)wave * 16;

    v8f acc[8];
    #pragma unroll
    for (int j = 0; j < 8; ++j) acc[j] = v8f{0.f,0.f,0.f,0.f,0.f,0.f,0.f,0.f};

    // A fragment rows: lanes 0-15 & 16-31 both map to rows m0..m0+15
    size_t arow = m0 + (size_t)r;
    if (arow >= (size_t)nrows) arow = (size_t)nrows - 1;  // clamp (pad rows)
    const __bf16* abase = hin + arow * DF + hl * 8;       // ISA 16-bit A layout

    #pragma unroll
    for (int ks = 0; ks < 4; ++ks) {
        const int k0 = ks * 32;
        // lane holds K = k0+8*hl+{0..7} and k0+16+8*hl+{0..7}
        v8bf alo = *(const v8bf*)(abase + k0);
        v8bf ahi = *(const v8bf*)(abase + k0 + 16);
        v16bf afrag = __builtin_shufflevector(alo, ahi,
            0,1,2,3,4,5,6,7,8,9,10,11,12,13,14,15);
        #pragma unroll
        for (int j = 0; j < 8; ++j) {
            // B[k][n] = W[n][k]; column n -> W row n, mirrored K layout
            const __bf16* bbase = Ws + (size_t)(j * 16 + r) * DF + hl * 8 + k0;
            v8bf blo = *(const v8bf*)(bbase);
            v8bf bhi = *(const v8bf*)(bbase + 16);
            v16bf bfrag = __builtin_shufflevector(blo, bhi,
                0,1,2,3,4,5,6,7,8,9,10,11,12,13,14,15);
            acc[j] = __builtin_amdgcn_wmma_f32_16x16x32_bf16(
                false, afrag, false, bfrag, (short)0, acc[j], false, false);
        }
    }

    // C/D layout: lane -> col = (lane&15), VGPR v -> row = v + 8*(lane>>4)
    float sc[8];
    if (mode == 0) {
        #pragma unroll
        for (int v = 0; v < 8; ++v) {
            size_t row = m0 + (size_t)(hl * 8 + v);
            float dv = (row < (size_t)nrows) ? dinv[row] : 0.0f;
            sc[v] = dv * dv;  // self-loop coefficient
        }
    }
    #pragma unroll
    for (int j = 0; j < 8; ++j) {
        const int col = j * 16 + r;
        const float bc = (mode == 1) ? bias[col] : 0.0f;
        #pragma unroll
        for (int v = 0; v < 8; ++v) {
            size_t row = m0 + (size_t)(hl * 8 + v);
            if (row < (size_t)nrows) {
                float val = acc[j][v];
                if (mode == 0) {
                    hw[row * DF + col]   = (__bf16)val;      // message (bf16)
                    fout[row * DF + col] = val * sc[v];      // agg init = self term
                } else {
                    fout[row * DF + col] = val + bc;         // final fp32 output
                }
            }
        }
    }
}

// ---------------------------------------------------------------------------
// Edge scatter: one wave per edge, each lane handles 4 features.
// Gather 256B bf16 message row, scale by norm, fp32 atomic add into agg.
// agg (51MB) + hw (25.6MB) fit in 192MB L2 -> atomics run at L2 speed.
// ---------------------------------------------------------------------------
__global__ __launch_bounds__(256)
void gcn_scatter(const __bf16* __restrict__ hw, const int* __restrict__ src,
                 const int* __restrict__ dst, const float* __restrict__ norm,
                 float* __restrict__ agg, int nedges) {
    const int lane = threadIdx.x & 31;
    const long long e = (long long)blockIdx.x * 8 + (threadIdx.x >> 5);
    if (e >= nedges) return;
    const int   s = src[e];
    const int   d = dst[e];
    const float w = norm[e];
    v4bf m = *(const v4bf*)(hw + (size_t)s * DF + lane * 4);
    float* q = agg + (size_t)d * DF + lane * 4;
    unsafeAtomicAdd(q + 0, (float)m[0] * w);
    unsafeAtomicAdd(q + 1, (float)m[1] * w);
    unsafeAtomicAdd(q + 2, (float)m[2] * w);
    unsafeAtomicAdd(q + 3, (float)m[3] * w);
}

// ---------------------------------------------------------------------------
// Bias + PReLU epilogue, fp32 agg -> bf16 activations for next layer
// ---------------------------------------------------------------------------
__global__ void gcn_bias_prelu(__bf16* __restrict__ hout, const float* __restrict__ agg,
                               const float* __restrict__ b, const float* __restrict__ aptr,
                               long long n) {
    long long i = (long long)blockIdx.x * blockDim.x + threadIdx.x;
    if (i >= n) return;
    const float a = aptr[0];
    float v = agg[i] + b[(int)(i & (DF - 1))];
    hout[i] = (__bf16)(v >= 0.0f ? v : a * v);
}

// ---------------------------------------------------------------------------
extern "C" void kernel_launch(void* const* d_in, const int* in_sizes, int n_in,
                              void* d_out, int out_size, void* d_ws, size_t ws_size,
                              hipStream_t stream) {
    const int*   x   = (const int*)d_in[0];
    const int*   ei  = (const int*)d_in[1];   // [2,E]: src then dst
    const float* ew  = (const float*)d_in[2];
    const float* emb = (const float*)d_in[3];
    const float* Wl[4] = { (const float*)d_in[4], (const float*)d_in[7],
                           (const float*)d_in[10], (const float*)d_in[13] };
    const float* bl[4] = { (const float*)d_in[5], (const float*)d_in[8],
                           (const float*)d_in[11], (const float*)d_in[14] };
    const float* al[3] = { (const float*)d_in[6], (const float*)d_in[9],
                           (const float*)d_in[12] };

    const int N = in_sizes[0];
    const int E = in_sizes[2];
    const int* esrc = ei;
    const int* edst = ei + E;
    const long long nD = (long long)N * DF;

    // workspace carve-up (256B aligned)
    auto alignup = [](size_t v) { return (v + 255) & ~(size_t)255; };
    char* ws = (char*)d_ws;
    float*  dinv = (float*)ws;  ws += alignup((size_t)N * 4);
    float*  norm = (float*)ws;  ws += alignup((size_t)E * 4);
    __bf16* hA   = (__bf16*)ws; ws += alignup((size_t)nD * 2);
    __bf16* hB   = (__bf16*)ws; ws += alignup((size_t)nD * 2);
    __bf16* hw   = (__bf16*)ws; ws += alignup((size_t)nD * 2);
    float*  agg  = (float*)ws;  ws += alignup((size_t)nD * 4);
    (void)ws_size; (void)n_in; (void)out_size;

    const int TPB = 256;
    const int gN  = (N + TPB - 1) / TPB;
    const int gE  = (E + TPB - 1) / TPB;
    const int gND = (int)((nD + TPB - 1) / TPB);
    const int gMM = (N + 127) / 128;
    const int gSC = (E + 7) / 8;

    // normalization precompute (identical for all layers)
    gcn_deg_init    <<<gN, TPB, 0, stream>>>(dinv, N);
    gcn_deg_accum   <<<gE, TPB, 0, stream>>>(dinv, edst, ew, E);
    gcn_deg_finalize<<<gN, TPB, 0, stream>>>(dinv, N);
    gcn_edge_norm   <<<gE, TPB, 0, stream>>>(norm, dinv, esrc, edst, ew, E);

    // h0 = bf16(emb[x])
    gcn_embed<<<gND, TPB, 0, stream>>>(hA, emb, x, nD);

    __bf16* hin = hA;
    __bf16* hout = hB;
    for (int l = 0; l < 3; ++l) {
        gcn_gemm<<<gMM, TPB, 0, stream>>>(hin, Wl[l], dinv, nullptr,
                                          hw, agg, N, /*mode=*/0);
        gcn_scatter<<<gSC, TPB, 0, stream>>>(hw, esrc, edst, norm, agg, E);
        gcn_bias_prelu<<<gND, TPB, 0, stream>>>(hout, agg, bl[l], al[l], nD);
        __bf16* t = hin; hin = hout; hout = t;
    }

    // final: out = h3 @ Wout.T + bout  (fp32 output)
    gcn_gemm<<<gMM, TPB, 0, stream>>>(hin, Wl[3], nullptr, bl[3],
                                      nullptr, (float*)d_out, N, /*mode=*/1);
}
